// GATPerformanceModel_36309653520735
// MI455X (gfx1250) — compile-verified, matrix operand order checked by reading, not run on hardware
//
#include <hip/hip_runtime.h>
#include <hip/hip_bf16.h>

// ---------------------------------------------------------------------------
// GAT model: gat1 (4 heads, concat) -> relu -> gat2 (1 head) -> relu ->
// global mean pool -> MLP.  N=10000, E=160000(+N loops), HID=128, G=64.
// The dense 10000x512x128 GEMM runs on v_wmma_f32_16x16x32_bf16.
// ---------------------------------------------------------------------------

typedef __attribute__((ext_vector_type(16))) __bf16 v16bf;
typedef __attribute__((ext_vector_type(8)))  float  v8f;

#define HID   128
#define H1    4
#define F1    512   // H1*HID
#define GRAPHS 64
#define OUTD  4
#define NEG   0.2f

__device__ __forceinline__ unsigned short f2bf(float f) {
    unsigned u = __float_as_uint(f);
    u = (u + 0x7FFFu + ((u >> 16) & 1u)) >> 16;   // round-to-nearest-even
    return (unsigned short)u;
}

__device__ __forceinline__ void atomicMaxF(float* a, float v) {
    unsigned int* ua = (unsigned int*)a;
    unsigned int old = *ua;
    while (__uint_as_float(old) < v) {
        unsigned int assumed = old;
        old = atomicCAS(ua, assumed, __float_as_uint(v));
        if (old == assumed) break;
    }
}

// ---------------------------------------------------------------------------
__global__ void k_fill(float* p, float v, int n) {
    int i = blockIdx.x * blockDim.x + threadIdx.x;
    if (i < n) p[i] = v;
}

// xl1[n, 512] = x[n, 0:6] @ W1[6, 512]   (bias b1 applied post-aggregation)
__global__ void k_lin1(const float* __restrict__ x, const float* __restrict__ W,
                       float* __restrict__ xl, int N) {
    int i = blockIdx.x * blockDim.x + threadIdx.x;
    if (i >= N * F1) return;
    int n = i >> 9, j = i & (F1 - 1);
    const float* xr = x + n * 6;
    float s = 0.f;
#pragma unroll
    for (int k = 0; k < 6; ++k) s += xr[k] * W[k * F1 + j];
    xl[i] = s;
}

// es[n,h] = <xl[n,h,:], a_src[h,:]>,  ed likewise
__global__ void k_attdot(const float* __restrict__ xl, const float* __restrict__ asrc,
                         const float* __restrict__ adst, float* __restrict__ es,
                         float* __restrict__ ed, int N, int H, int C) {
    int i = blockIdx.x * blockDim.x + threadIdx.x;
    if (i >= N * H) return;
    int n = i / H, h = i % H;
    const float* row = xl + (size_t)n * H * C + h * C;
    float s = 0.f, d = 0.f;
    for (int c = 0; c < C; ++c) {
        float v = row[c];
        s += v * asrc[h * C + c];
        d += v * adst[h * C + c];
    }
    es[i] = s;
    ed[i] = d;
}

__device__ __forceinline__ void edge_sd(const long long* ei, int e, int E, int& s, int& d) {
    if (e < E) { s = (int)ei[e]; d = (int)ei[E + e]; }
    else       { s = e - E;      d = s; }
}

// pass 1: e = leaky_relu(es[src]+ed[dst]); segment max into m[dst]
__global__ void k_edge_max(const long long* __restrict__ ei, const float* __restrict__ es,
                           const float* __restrict__ ed, float* __restrict__ ebuf,
                           float* __restrict__ m, int E, int EP, int H) {
    int i = blockIdx.x * blockDim.x + threadIdx.x;
    if (i >= EP * H) return;
    int e = i / H, h = i % H, s, d;
    edge_sd(ei, e, E, s, d);
    float v = es[s * H + h] + ed[d * H + h];
    v = v > 0.f ? v : NEG * v;
    ebuf[i] = v;
    atomicMaxF(&m[d * H + h], v);
}

// pass 2: ex = exp(e - m[dst]); segment sum into den[dst]
__global__ void k_edge_exp(const long long* __restrict__ ei, float* __restrict__ ebuf,
                           const float* __restrict__ m, float* __restrict__ den,
                           int E, int EP, int H) {
    int i = blockIdx.x * blockDim.x + threadIdx.x;
    if (i >= EP * H) return;
    int e = i / H, h = i % H, s, d;
    edge_sd(ei, e, E, s, d);
    float v = __expf(ebuf[i] - m[d * H + h]);
    ebuf[i] = v;
    atomicAdd(&den[d * H + h], v);
}

// pass 3: alpha = ex / den[dst]
__global__ void k_edge_alpha(const long long* __restrict__ ei, float* __restrict__ ebuf,
                             const float* __restrict__ den, int E, int EP, int H) {
    int i = blockIdx.x * blockDim.x + threadIdx.x;
    if (i >= EP * H) return;
    int e = i / H, h = i % H, s, d;
    edge_sd(ei, e, E, s, d);
    ebuf[i] = ebuf[i] / den[d * H + h];
}

// pass 4: out[dst, :] += alpha * xl[src, :]
__global__ void k_agg(const long long* __restrict__ ei, const float* __restrict__ ebuf,
                      const float* __restrict__ xl, float* __restrict__ out,
                      int E, int EP, int H, int C) {
    long long i = (long long)blockIdx.x * blockDim.x + threadIdx.x;
    int HC = H * C;
    if (i >= (long long)EP * HC) return;
    int e = (int)(i / HC), j = (int)(i % HC), s, d;
    edge_sd(ei, e, E, s, d);
    float a = ebuf[e * H + (j / C)];
    atomicAdd(&out[(size_t)d * HC + j], a * xl[(size_t)s * HC + j]);
}

// h = relu(out1 + b1) rounded to bf16 (feeds only the W2 GEMM)
__global__ void k_h1bf(const float* __restrict__ out1, const float* __restrict__ b1,
                       unsigned short* __restrict__ hbf, int N) {
    int i = blockIdx.x * blockDim.x + threadIdx.x;
    if (i >= N * F1) return;
    float v = out1[i] + b1[i & (F1 - 1)];
    hbf[i] = f2bf(v > 0.f ? v : 0.f);
}

// Pack W2[512,128] into per-lane WMMA B fragments: [kt(16)][nt(8)][lane(32)][16]
__global__ void k_packB(const float* __restrict__ W2, unsigned short* __restrict__ Bp) {
    int i = blockIdx.x * blockDim.x + threadIdx.x;
    if (i >= 16 * 8 * 32 * 16) return;
    int idx  = i & 15;
    int lane = (i >> 4) & 31;
    int nt   = (i >> 9) & 7;
    int kt   = i >> 12;
    int g    = lane >> 4;
    int n    = nt * 16 + (lane & 15);
    int k    = kt * 32 + (idx < 8 ? g * 8 + idx : 16 + g * 8 + (idx - 8));
    Bp[i] = f2bf(W2[k * HID + n]);
}

// xl2[M,128] = hbf[M,512] @ W2bf[512,128]  via v_wmma_f32_16x16x32_bf16.
// One wave per 16x16 tile; wave w of the block owns N-tile w; block = M-tile.
__global__ __launch_bounds__(256)
void k_gemm_wmma(const unsigned short* __restrict__ Abf,
                 const unsigned short* __restrict__ Bp,
                 float* __restrict__ C, int M) {
    int wave = threadIdx.x >> 5;        // N tile 0..7
    int lane = threadIdx.x & 31;
    int mt   = blockIdx.x;
    int g    = lane >> 4;               // K half-group
    int mrow = mt * 16 + (lane & 15);
    int mrl  = mrow < M ? mrow : M - 1; // clamp loads at the tail
    const unsigned short* arow = Abf + (size_t)mrl * F1;

    v8f acc = {};
    for (int kt = 0; kt < 16; ++kt) {
        union { uint4 u[2]; v16bf v; } a, b;
        a.u[0] = *(const uint4*)(arow + kt * 32 + g * 8);
        a.u[1] = *(const uint4*)(arow + kt * 32 + 16 + g * 8);
        const unsigned short* bp = Bp + (((kt * 8 + wave) * 32 + lane) << 4);
        b.u[0] = *(const uint4*)(bp);
        b.u[1] = *(const uint4*)(bp + 8);
        acc = __builtin_amdgcn_wmma_f32_16x16x32_bf16(
            false, a.v, false, b.v, (short)0, acc, false, false);
    }
    int nCol  = wave * 16 + (lane & 15);
    int mBase = mt * 16 + ((lane >> 4) << 3);
#pragma unroll
    for (int v = 0; v < 8; ++v) {
        int m = mBase + v;
        if (m < M) C[(size_t)m * HID + nCol] = acc[v];
    }
}

// pool: relu(out2 + b2) scattered into per-graph sums (+ node counts)
__global__ void k_pool(const float* __restrict__ out2, const float* __restrict__ b2,
                       const long long* __restrict__ batch, float* __restrict__ pooled,
                       float* __restrict__ cnt, int N) {
    int i = blockIdx.x * blockDim.x + threadIdx.x;
    if (i >= N * HID) return;
    int n = i >> 7, c = i & (HID - 1);
    float v = out2[i] + b2[c];
    v = v > 0.f ? v : 0.f;
    int gidx = (int)batch[n];
    atomicAdd(&pooled[gidx * HID + c], v);
    if (c == 0) atomicAdd(&cnt[gidx], 1.0f);
}

__global__ void k_poolfin(float* pooled, const float* cnt) {
    int i = blockIdx.x * blockDim.x + threadIdx.x;
    if (i >= GRAPHS * HID) return;
    pooled[i] /= fmaxf(cnt[i >> 7], 1.0f);
}

__global__ void k_mlp1(const float* __restrict__ pooled, const float* __restrict__ mw1,
                       const float* __restrict__ mb1, float* __restrict__ t1) {
    int i = blockIdx.x * blockDim.x + threadIdx.x;
    if (i >= GRAPHS * HID) return;
    int gr = i >> 7, j = i & (HID - 1);
    float s = mb1[j];
    for (int k = 0; k < HID; ++k) s += pooled[gr * HID + k] * mw1[k * HID + j];
    t1[i] = s > 0.f ? s : 0.f;
}

__global__ void k_mlp2(const float* __restrict__ t1, const float* __restrict__ mw2,
                       const float* __restrict__ mb2, float* __restrict__ out) {
    int i = blockIdx.x * blockDim.x + threadIdx.x;
    if (i >= GRAPHS * OUTD) return;
    int gr = i >> 2, o = i & (OUTD - 1);
    float s = mb2[o];
    for (int j = 0; j < HID; ++j) s += t1[gr * HID + j] * mw2[j * OUTD + o];
    out[i] = s;
}

// ---------------------------------------------------------------------------
extern "C" void kernel_launch(void* const* d_in, const int* in_sizes, int n_in,
                              void* d_out, int out_size, void* d_ws, size_t ws_size,
                              hipStream_t stream) {
    const float*     x      = (const float*)d_in[0];
    const long long* ei     = (const long long*)d_in[1];
    const long long* batch  = (const long long*)d_in[2];
    const float*     W1     = (const float*)d_in[3];
    const float*     asrc1  = (const float*)d_in[4];
    const float*     adst1  = (const float*)d_in[5];
    const float*     b1     = (const float*)d_in[6];
    const float*     W2     = (const float*)d_in[7];
    const float*     asrc2  = (const float*)d_in[8];
    const float*     adst2  = (const float*)d_in[9];
    const float*     b2     = (const float*)d_in[10];
    const float*     mw1    = (const float*)d_in[11];
    const float*     mb1    = (const float*)d_in[12];
    const float*     mw2    = (const float*)d_in[13];
    const float*     mb2    = (const float*)d_in[14];
    float*           out    = (float*)d_out;

    const int N  = in_sizes[0] / 6;
    const int E  = in_sizes[1] / 2;
    const int EP = E + N;

    // ---- workspace layout (256B-aligned regions, with reuse) ----
    char* ws = (char*)d_ws;
    auto alloc = [&](size_t bytes) -> char* {
        char* p = ws;
        ws += (bytes + 255) & ~(size_t)255;
        return p;
    };
    float* xl1   = (float*)alloc((size_t)N * F1 * 4);  // reused as hbf (u16)
    float* out1  = (float*)alloc((size_t)N * F1 * 4);  // reused as xl2/out2
    float* es1   = (float*)alloc((size_t)N * H1 * 4);
    float* ed1   = (float*)alloc((size_t)N * H1 * 4);
    float* m1    = (float*)alloc((size_t)N * H1 * 4);
    float* den1  = (float*)alloc((size_t)N * H1 * 4);
    float* eb1   = (float*)alloc((size_t)EP * H1 * 4);
    unsigned short* Bp = (unsigned short*)alloc(16 * 8 * 32 * 16 * 2);
    float* es2   = (float*)alloc((size_t)N * 4);
    float* ed2   = (float*)alloc((size_t)N * 4);
    float* m2    = (float*)alloc((size_t)N * 4);
    float* den2  = (float*)alloc((size_t)N * 4);
    float* eb2   = (float*)alloc((size_t)EP * 4);
    float* pooled= (float*)alloc(GRAPHS * HID * 4);
    float* cnt   = (float*)alloc(GRAPHS * 4);
    float* t1    = (float*)alloc(GRAPHS * HID * 4);
    unsigned short* hbf = (unsigned short*)xl1;        // alias (xl1 dead by then)
    float* xl2   = out1;                               // alias (out1 dead by then)
    float* out2  = out1 + (size_t)N * HID;

    const int T = 256;
    auto blk = [](long long n) { return (int)((n + 255) / 256); };

    // ---- init ----
    k_fill<<<blk((long long)N * F1), T, 0, stream>>>(out1, 0.f, N * F1);
    k_fill<<<blk(N * H1), T, 0, stream>>>(m1, -3.0e38f, N * H1);
    k_fill<<<blk(N * H1), T, 0, stream>>>(den1, 0.f, N * H1);
    k_fill<<<blk(N), T, 0, stream>>>(m2, -3.0e38f, N);
    k_fill<<<blk(N), T, 0, stream>>>(den2, 0.f, N);
    k_fill<<<blk(GRAPHS * HID), T, 0, stream>>>(pooled, 0.f, GRAPHS * HID);
    k_fill<<<1, T, 0, stream>>>(cnt, 0.f, GRAPHS);

    // ---- gat1 ----
    k_lin1<<<blk((long long)N * F1), T, 0, stream>>>(x, W1, xl1, N);
    k_attdot<<<blk(N * H1), T, 0, stream>>>(xl1, asrc1, adst1, es1, ed1, N, H1, HID);
    k_edge_max  <<<blk((long long)EP * H1), T, 0, stream>>>(ei, es1, ed1, eb1, m1, E, EP, H1);
    k_edge_exp  <<<blk((long long)EP * H1), T, 0, stream>>>(ei, eb1, m1, den1, E, EP, H1);
    k_edge_alpha<<<blk((long long)EP * H1), T, 0, stream>>>(ei, eb1, den1, E, EP, H1);
    k_agg<<<blk((long long)EP * F1), T, 0, stream>>>(ei, eb1, xl1, out1, E, EP, H1, HID);
    k_h1bf<<<blk((long long)N * F1), T, 0, stream>>>(out1, b1, hbf, N);

    // ---- gat2 linear: WMMA bf16 ----
    k_packB<<<blk(16 * 8 * 32 * 16), T, 0, stream>>>(W2, Bp);
    k_gemm_wmma<<<(N + 15) / 16, 256, 0, stream>>>(hbf, Bp, xl2, N);

    // ---- gat2 attention/aggregation (note: out2 must be zeroed after xl2 write
    //      since it aliases the second quarter of the out1 region) ----
    k_fill<<<blk((long long)N * HID), T, 0, stream>>>(out2, 0.f, N * HID);
    k_attdot<<<blk(N), T, 0, stream>>>(xl2, asrc2, adst2, es2, ed2, N, 1, HID);
    k_edge_max  <<<blk(EP), T, 0, stream>>>(ei, es2, ed2, eb2, m2, E, EP, 1);
    k_edge_exp  <<<blk(EP), T, 0, stream>>>(ei, eb2, m2, den2, E, EP, 1);
    k_edge_alpha<<<blk(EP), T, 0, stream>>>(ei, eb2, den2, E, EP, 1);
    k_agg<<<blk((long long)EP * HID), T, 0, stream>>>(ei, eb2, xl2, out2, E, EP, 1, HID);

    // ---- pool + MLP ----
    k_pool<<<blk((long long)N * HID), T, 0, stream>>>(out2, b2, batch, pooled, cnt, N);
    k_poolfin<<<blk(GRAPHS * HID), T, 0, stream>>>(pooled, cnt);
    k_mlp1<<<blk(GRAPHS * HID), T, 0, stream>>>(pooled, mw1, mb1, t1);
    k_mlp2<<<1, 256, 0, stream>>>(t1, mw2, mb2, out);
}
